// NERLSTM_CRF_29489245454934
// MI455X (gfx1250) — matrix operand order repaired and checked by reading, not compiled, not run on hardware
//
#include <hip/hip_runtime.h>
#include <hip/hip_bf16.h>

typedef __attribute__((ext_vector_type(16))) _Float16 v16h;
typedef __attribute__((ext_vector_type(8)))  float    v8f;

constexpr int Bsz  = 512;   // batch
constexpr int Tsz  = 200;   // seq len
constexpr int CDv  = 100;   // char emb dim
constexpr int LDv  = 20;    // len emb dim
constexpr int Ev   = 120;   // E = CD + LD
constexpr int Hv   = 128;   // hidden per direction
constexpr int G4v  = 512;   // 4*H gates
constexpr int Kv   = 13;    // tags
constexpr int KTn  = 4;     // k-tiles of 32 (E padded to 128, H = 128)
constexpr int NTn  = 32;    // n-tiles of 16 (512/16)
constexpr int RTn  = 16;    // batch rows per block (WMMA M)
constexpr int NBLK = Bsz / RTn;          // 32 blocks per direction
constexpr int WFRAG = NTn * KTn * 512;   // halves per weight matrix = 65536

// ---------------------------------------------------------------------------
// Swizzle (CDNA5 ISA §7.12.2, 16-bit layouts, wave32)
// A 16x32 f16: (lane,s) -> m=lane&15, k=(s&7)+8*(lane>>4)+16*(s>>3)
// B 32x16 f16: (lane,s) -> n=lane&15, k=(s&7)+8*(lane>>4)+16*(s>>3)
// C 16x16 f32: VGPR r, lane l -> M = r + 8*(l>>4), N = l&15
// ---------------------------------------------------------------------------

// CDNA5 async copy global->LDS (CDNA5 ISA):
// one B128 per lane (16B); tracked by ASYNCcnt.
__device__ __forceinline__ void async_copy_b128(unsigned lds_byte_off, const void* gaddr) {
    asm volatile("global_load_async_to_lds_b128 %0, %1, off"
                 :: "v"(lds_byte_off), "v"((unsigned long long)(size_t)gaddr)
                 : "memory");
}
__device__ __forceinline__ void wait_async0() {
    asm volatile("s_wait_asynccnt 0x0" ::: "memory");
}
// Low 32 bits of a flat shared-space address are the LDS byte offset (ISA §10.2).
__device__ __forceinline__ unsigned lds_off(const void* p) {
    return (unsigned)(size_t)p;
}

// Kernel 1: weights -> f16 swizzled B-fragments (4 matrices) + fused biases
__global__ __launch_bounds__(256)
void prep_weights(const float* __restrict__ Wih_f, const float* __restrict__ Whh_f,
                  const float* __restrict__ Wih_b, const float* __restrict__ Whh_b,
                  const float* __restrict__ bih_f, const float* __restrict__ bhh_f,
                  const float* __restrict__ bih_b, const float* __restrict__ bhh_b,
                  _Float16* __restrict__ wb, float* __restrict__ bias)
{
    int idx = blockIdx.x * 256 + threadIdx.x;
    if (idx < 4 * WFRAG) {
        int mi = idx / WFRAG;          // 0:Wih_f 1:Whh_f 2:Wih_b 3:Whh_b
        int r  = idx % WFRAG;
        int ntile = r / (KTn * 512);
        int r2    = r % (KTn * 512);
        int kt    = r2 / 512;
        int li    = r2 % 512;
        int lane = li >> 4, s = li & 15;
        int nloc = lane & 15;
        int kk   = (s & 7) + 8 * (lane >> 4) + 16 * (s >> 3);     // B layout
        int k    = kt * 32 + kk;
        int n    = ntile * 16 + nloc;
        float v = 0.f;
        if (mi == 0)      { if (k < Ev) v = Wih_f[n * Ev + k]; }
        else if (mi == 1) { v = Whh_f[n * Hv + k]; }
        else if (mi == 2) { if (k < Ev) v = Wih_b[n * Ev + k]; }
        else              { v = Whh_b[n * Hv + k]; }
        wb[idx] = (_Float16)v;
    } else if (idx < 4 * WFRAG + 2 * G4v) {
        int j = idx - 4 * WFRAG;
        if (j < G4v) bias[j] = bih_f[j] + bhh_f[j];
        else         bias[j] = bih_b[j - G4v] + bhh_b[j - G4v];
    }
}

// Kernel 2: embedding gather directly into swizzled f16 A-fragments
__global__ __launch_bounds__(256)
void embed_swizzle(const int* __restrict__ char_ids, const int* __restrict__ len_ids,
                   const float* __restrict__ char_emb, const float* __restrict__ len_emb,
                   _Float16* __restrict__ embA)
{
    long long idx = (long long)blockIdx.x * 256 + threadIdx.x;
    const long long total = (long long)NBLK * Tsz * KTn * 512;
    if (idx >= total) return;
    int li = (int)(idx & 511);
    long long q = idx >> 9;
    int kt = (int)(q & 3); q >>= 2;
    int t  = (int)(q % Tsz);
    int rt = (int)(q / Tsz);
    int lane = li >> 4, s = li & 15;
    int m  = lane & 15, hi = lane >> 4;
    int kk = (s & 7) + 8 * hi + 16 * (s >> 3);   // A layout inverse
    int k  = kt * 32 + kk;
    int b  = rt * RTn + m;
    float v = 0.f;
    if (k < CDv)      v = char_emb[(long long)char_ids[b * Tsz + t] * CDv + k];
    else if (k < Ev)  v = len_emb[len_ids[b * Tsz + t] * LDv + (k - CDv)];
    embA[idx] = (_Float16)v;
}

// Kernel 3: persistent per-batch-tile LSTM (one direction).
// 256 threads = 8 waves. Wave w takes n-tile (g*8 + w) for gate g = 0..3,
// so acc[0..3] = i,f,g,o for the SAME columns col = w*16 + (lane&15) and
// rows m = r + 8*(lane>>4): the whole cell update runs in registers.
// Weights register-resident; x(t+1) async-prefetched to LDS; h double-buffered
// in LDS A-layout -> exactly ONE barrier per timestep.
__global__ __launch_bounds__(256)
void lstm_dir(const _Float16* __restrict__ embA,
              const _Float16* __restrict__ wihB,
              const _Float16* __restrict__ whhB,
              const float* __restrict__ bias,
              float* __restrict__ hs, int reverse)
{
    __shared__ alignas(32) _Float16 xbuf[2][KTn * 512]; // 8KB double-buffered x frags
    __shared__ alignas(32) _Float16 hAb[2][KTn * 512];  // 8KB double-buffered h frags

    const int tid  = threadIdx.x;
    const int lane = tid & 31;
    const int wave = tid >> 5;
    const int rt   = blockIdx.x;
    const int n0   = lane & 15;
    const int hi16 = lane >> 4;
    const int col  = wave * 16 + n0;      // within-gate column owned by this lane

    // per-lane gate biases (i,f,g,o)
    float bg[4];
    #pragma unroll
    for (int g = 0; g < 4; ++g) bg[g] = bias[g * Hv + col];

    // Preload weight fragments: acc[g] <- n-tile g*8 + wave
    v16h wih[KTn][4], whh[KTn][4];
    #pragma unroll
    for (int kt = 0; kt < KTn; ++kt) {
        #pragma unroll
        for (int g = 0; g < 4; ++g) {
            const int nt = g * 8 + wave;
            wih[kt][g] = *(const v16h*)(wihB + (nt * KTn + kt) * 512 + lane * 16);
            whh[kt][g] = *(const v16h*)(whhB + (nt * KTn + kt) * 512 + lane * 16);
        }
    }

    // zero h(0) buffer
    for (int i = tid; i < KTn * 512; i += 256) hAb[0][i] = (_Float16)0.f;

    // cell state for this lane's 8 cells (rows r+8*hi16, column col)
    float creg[8] = {0.f, 0.f, 0.f, 0.f, 0.f, 0.f, 0.f, 0.f};

    // Each wave async-copies its 512B chunk of the 4KB x-fragment block.
    auto issue_x = [&](int t, int buf) {
        const _Float16* gp = embA + ((long long)(rt * Tsz + t) * KTn) * 512
                                  + wave * 256 + lane * 8;            // halves
        unsigned lo = lds_off(&xbuf[buf][wave * 256 + lane * 8]);
        async_copy_b128(lo, gp);
    };

    issue_x(reverse ? (Tsz - 1) : 0, 0);
    wait_async0();
    __syncthreads();

    // h A-layout coordinates for this lane's column (constant over the loop)
    const int ktc = col >> 5, kkc = col & 31;
    const int hic = ((kkc & 15) >= 8) ? 1 : 0;
    const int sc  = (kkc & 7) + 8 * (kkc >> 4);

    int cur = 0;
    for (int step = 0; step < Tsz; ++step) {
        const int t = reverse ? (Tsz - 1 - step) : step;
        if (step + 1 < Tsz) {
            const int tn = reverse ? (Tsz - 2 - step) : (step + 1);
            issue_x(tn, cur ^ 1);
        }

        // A fragments from LDS, then a clean 32-WMMA chain
        v16h ax[KTn], ah[KTn];
        #pragma unroll
        for (int kt = 0; kt < KTn; ++kt)
            ax[kt] = *(const v16h*)(&xbuf[cur][kt * 512 + lane * 16]);
        #pragma unroll
        for (int kt = 0; kt < KTn; ++kt)
            ah[kt] = *(const v16h*)(&hAb[cur][kt * 512 + lane * 16]);

        v8f acc[4] = {};
        #pragma unroll
        for (int kt = 0; kt < KTn; ++kt) {
            #pragma unroll
            for (int g = 0; g < 4; ++g)
                acc[g] = __builtin_amdgcn_wmma_f32_16x16x32_f16(
                    false, ax[kt], false, wih[kt][g], (short)0, acc[g], false, false);
        }
        #pragma unroll
        for (int kt = 0; kt < KTn; ++kt) {
            #pragma unroll
            for (int g = 0; g < 4; ++g)
                acc[g] = __builtin_amdgcn_wmma_f32_16x16x32_f16(
                    false, ah[kt], false, whh[kt][g], (short)0, acc[g], false, false);
        }

        // LSTM cell entirely in registers off the C-layout accumulators
        _Float16* hAn = &hAb[cur ^ 1][0];
        #pragma unroll
        for (int r = 0; r < 8; ++r) {
            const int m = r + 8 * hi16;
            float iv = acc[0][r] + bg[0];
            float fv = acc[1][r] + bg[1];
            float gv = acc[2][r] + bg[2];
            float ov = acc[3][r] + bg[3];
            float si = 1.f / (1.f + __expf(-iv));
            float sf = 1.f / (1.f + __expf(-fv));
            float so = 1.f / (1.f + __expf(-ov));
            float c  = sf * creg[r] + si * tanhf(gv);
            float h  = so * tanhf(c);
            creg[r] = c;
            hs[((long long)(rt * RTn + m) * Tsz + t) * Hv + col] = h;
            hAn[ktc * 512 + (m + 16 * hic) * 16 + sc] = (_Float16)h;
        }

        wait_async0();      // retire this wave's x(t+1) copy
        __syncthreads();    // publish h(t) and x(t+1) for next step
        cur ^= 1;
    }
}

// Kernel 4: emissions = [hf|hb] @ W_out^T + b_out   (K=13, VALU is fine)
__global__ __launch_bounds__(256)
void emissions_k(const float* __restrict__ hf, const float* __restrict__ hb,
                 const float* __restrict__ W_out, const float* __restrict__ b_out,
                 float* __restrict__ em)
{
    long long gid = (long long)blockIdx.x * 256 + threadIdx.x;
    const long long total = (long long)Bsz * Tsz * Kv;
    if (gid >= total) return;
    long long bt = gid / Kv;
    int k = (int)(gid % Kv);
    const float* hfp = hf + bt * Hv;
    const float* hbp = hb + bt * Hv;
    const float* wo  = W_out + (long long)k * (2 * Hv);
    float s = b_out[k];
    #pragma unroll 8
    for (int j = 0; j < Hv; ++j) s += hfp[j] * wo[j];
    #pragma unroll 8
    for (int j = 0; j < Hv; ++j) s += hbp[j] * wo[Hv + j];
    em[gid] = s;
}

// Kernel 5: Viterbi decode, one thread per batch row (K=13)
__global__ __launch_bounds__(256)
void viterbi_k(const float* __restrict__ em, const unsigned char* __restrict__ mask,
               const float* __restrict__ startT, const float* __restrict__ endT,
               const float* __restrict__ trans,
               unsigned char* __restrict__ hist, float* __restrict__ out)
{
    __shared__ float sT[Kv * Kv], sS[Kv], sE[Kv];
    int tid = threadIdx.x;
    if (tid < Kv * Kv) sT[tid] = trans[tid];
    if (tid < Kv) { sS[tid] = startT[tid]; sE[tid] = endT[tid]; }
    __syncthreads();
    int b = blockIdx.x * 256 + tid;
    if (b >= Bsz) return;

    const float* em0 = em + (long long)b * Tsz * Kv;
    float score[Kv];
    #pragma unroll
    for (int k = 0; k < Kv; ++k) score[k] = sS[k] + em0[k];

    for (int t = 1; t < Tsz; ++t) {
        float emt[Kv];
        #pragma unroll
        for (int k = 0; k < Kv; ++k) emt[k] = em0[t * Kv + k];
        const bool mk = mask[b * Tsz + t] != 0;
        float ns[Kv]; int am[Kv];
        #pragma unroll
        for (int cur = 0; cur < Kv; ++cur) {
            float best = score[0] + sT[0 * Kv + cur]; int bi = 0;
            #pragma unroll
            for (int p = 1; p < Kv; ++p) {
                float v = score[p] + sT[p * Kv + cur];
                if (v > best) { best = v; bi = p; }   // first-max like jnp.argmax
            }
            ns[cur] = best + emt[cur];
            am[cur] = bi;
        }
        #pragma unroll
        for (int cur = 0; cur < Kv; ++cur) {
            if (mk) score[cur] = ns[cur];
            hist[((long long)b * (Tsz - 1) + (t - 1)) * 16 + cur] = (unsigned char)am[cur];
        }
    }
    #pragma unroll
    for (int k = 0; k < Kv; ++k) score[k] += sE[k];
    int tag = 0; float bs = score[0];
    #pragma unroll
    for (int k = 1; k < Kv; ++k) if (score[k] > bs) { bs = score[k]; tag = k; }
    out[(long long)b * Tsz + (Tsz - 1)] = (float)tag;
    for (int t = Tsz - 2; t >= 0; --t) {
        int prev = hist[((long long)b * (Tsz - 1) + t) * 16 + tag];
        if (mask[b * Tsz + (t + 1)] != 0) tag = prev;
        out[(long long)b * Tsz + t] = (float)tag;
    }
}

// ---------------------------------------------------------------------------
extern "C" void kernel_launch(void* const* d_in, const int* in_sizes, int n_in,
                              void* d_out, int out_size, void* d_ws, size_t ws_size,
                              hipStream_t stream)
{
    const int*   char_ids = (const int*)d_in[0];
    const int*   len_ids  = (const int*)d_in[1];
    const unsigned char* mask = (const unsigned char*)d_in[2]; // numpy bool = 1 byte
    const float* char_emb = (const float*)d_in[3];
    const float* len_emb  = (const float*)d_in[4];
    const float* Wih_f = (const float*)d_in[5];
    const float* Whh_f = (const float*)d_in[6];
    const float* bih_f = (const float*)d_in[7];
    const float* bhh_f = (const float*)d_in[8];
    const float* Wih_b = (const float*)d_in[9];
    const float* Whh_b = (const float*)d_in[10];
    const float* bih_b = (const float*)d_in[11];
    const float* bhh_b = (const float*)d_in[12];
    const float* W_out = (const float*)d_in[13];
    const float* b_out = (const float*)d_in[14];
    const float* startT = (const float*)d_in[15];
    const float* endT   = (const float*)d_in[16];
    const float* trans  = (const float*)d_in[17];
    (void)in_sizes; (void)n_in; (void)out_size; (void)ws_size;

    char* ws = (char*)d_ws;
    size_t off = 0;
    auto take = [&](size_t bytes) -> void* {
        void* p = ws + off;
        off = (off + bytes + 255) & ~size_t(255);
        return p;
    };
    _Float16* wb   = (_Float16*)take((size_t)4 * WFRAG * sizeof(_Float16));   // 512KB
    float*    bias = (float*)take((size_t)2 * G4v * sizeof(float));           // 4KB
    _Float16* embA = (_Float16*)take((size_t)NBLK * Tsz * KTn * 512 * 2);     // 25MB
    float*    hf   = (float*)take((size_t)Bsz * Tsz * Hv * sizeof(float));    // 50MB
    float*    hb   = (float*)take((size_t)Bsz * Tsz * Hv * sizeof(float));    // 50MB
    float*    em   = (float*)take((size_t)Bsz * Tsz * Kv * sizeof(float));    // 5.1MB
    unsigned char* hist = (unsigned char*)take((size_t)Bsz * (Tsz - 1) * 16); // 1.6MB

    prep_weights<<<1028, 256, 0, stream>>>(Wih_f, Whh_f, Wih_b, Whh_b,
                                           bih_f, bhh_f, bih_b, bhh_b, wb, bias);
    embed_swizzle<<<51200, 256, 0, stream>>>(char_ids, len_ids, char_emb, len_emb, embA);
    lstm_dir<<<NBLK, 256, 0, stream>>>(embA, wb, wb + WFRAG, bias, hf, 0);
    lstm_dir<<<NBLK, 256, 0, stream>>>(embA, wb + 2 * WFRAG, wb + 3 * WFRAG,
                                       bias + G4v, hb, 1);
    emissions_k<<<5200, 256, 0, stream>>>(hf, hb, W_out, b_out, em);
    viterbi_k<<<2, 256, 0, stream>>>(em, mask, startT, endT, trans, hist, (float*)d_out);
}
